// DeformableConv2D_91250875171198
// MI455X (gfx1250) — matrix operand order repaired and checked by reading, not compiled, not run on hardware
//
#include <hip/hip_runtime.h>
#include <math.h>

// ---------------------------------------------------------------------------
// DCNv2: B=8, H=64, W=64, C=128, F=256, KH=KW=3 (K=9), DG=1, SAME pad, stride 1
// ---------------------------------------------------------------------------
#define BN 8
#define HH 64
#define WW 64
#define CC 128
#define FF 256
#define KK 9
#define KC (KK * CC)        // 1152
#define NPIX (BN * HH * WW) // 32768

typedef __attribute__((ext_vector_type(16))) __bf16 v16bf;
typedef __attribute__((ext_vector_type(8)))  float  v8f;

union FragBF {
    v16bf v;
    uint4 q[2];
};

__device__ __forceinline__ unsigned short f2bf(float f) {
    union { float f; unsigned u; } a; a.f = f;
    unsigned r = a.u + 0x7FFFu + ((a.u >> 16) & 1u);   // round-to-nearest-even
    return (unsigned short)(r >> 16);
}

// ---------------------------------------------------------------------------
// Kernel 1: offset conv  (plain 3x3 SAME conv: C=128 -> 27 channels)
// ofs layout: [row][27]  (j<9: dy, 9..17: dx, 18..26: mask logits)
// ---------------------------------------------------------------------------
__global__ __launch_bounds__(256) void dcn_offset_conv(
    const float* __restrict__ in, const float* __restrict__ okern,
    const float* __restrict__ obias, float* __restrict__ ofs)
{
    int gid = blockIdx.x * 256 + threadIdx.x;
    if (gid >= NPIX * 27) return;
    int row = gid / 27;
    int j   = gid - row * 27;
    int b = row >> 12;
    int h = (row >> 6) & 63;
    int w = row & 63;

    float acc = obias[j];
    #pragma unroll
    for (int ky = 0; ky < 3; ++ky) {
        int y = h - 1 + ky;
        if (y < 0 || y >= HH) continue;
        #pragma unroll
        for (int kx = 0; kx < 3; ++kx) {
            int x = w - 1 + kx;
            if (x < 0 || x >= WW) continue;
            const float* ip = in + ((size_t)((b * HH + y) * WW + x)) * CC;
            const float* wp = okern + ((size_t)((ky * 3 + kx) * CC)) * 27 + j;
            #pragma unroll 4
            for (int c = 0; c < CC; ++c)
                acc = fmaf(ip[c], wp[(size_t)c * 27], acc);
        }
    }
    ofs[gid] = acc;
}

// ---------------------------------------------------------------------------
// Kernel 2: filter f32 [K][C][F] -> bf16 transposed [F][K*C]
// ---------------------------------------------------------------------------
__global__ __launch_bounds__(256) void dcn_filt_cvt(
    const float* __restrict__ filt, unsigned short* __restrict__ filtT)
{
    int gid = blockIdx.x * 256 + threadIdx.x;
    if (gid >= FF * KC) return;
    int f  = gid / KC;
    int kc = gid - f * KC;
    filtT[gid] = f2bf(filt[(size_t)kc * FF + f]);
}

// ---------------------------------------------------------------------------
// Kernel 3: fused modulated bilinear sampling + bf16 WMMA GEMM
// Block: 512 threads (16 waves). Block tile: M=128 pixels x N=256 filters.
// K loop: 36 slices of 32 (one k-tap, 32 channels per slice).
// Wave tile: 32x64 = 2x4 subtiles of v_wmma_f32_16x16x32_bf16.
// B tile staged with async global->LDS (ASYNCcnt path) while sampling runs.
// ---------------------------------------------------------------------------
#define LDA 40   // padded LDS row stride (bf16 elems) for A (128 x 32 tile)
#define LDB 40   // padded LDS col stride for B (column-major, 256 cols x 32 K)

__global__ __launch_bounds__(512) void dcn_gemm(
    const float* __restrict__ in, const float* __restrict__ ofs,
    const unsigned short* __restrict__ filtT, float* __restrict__ out)
{
    __shared__ unsigned short Ash[128 * LDA];  // 10240 B
    __shared__ unsigned short Bsh[256 * LDB];  // 20480 B

    const int tid    = threadIdx.x;
    const int lane   = tid & 31;
    const int waveId = tid >> 5;          // 0..15
    const int wm     = waveId & 3;        // M wave index -> rows wm*32
    const int wn     = waveId >> 2;       // N wave index -> cols wn*64
    const int blockM = blockIdx.x * 128;

    // --- per-thread sampling role: 128 rows x 4 channel-groups of 8 ---------
    const int rowL = tid >> 2;            // 0..127
    const int cg   = tid & 3;             // 0..3
    const int row  = blockM + rowL;
    const int b = row >> 12;
    const int h = (row >> 6) & 63;
    const int w = row & 63;
    const float* orow = ofs + (size_t)row * 27;

    // --- per-thread B staging role: 256 cols x 2 K-halves of 16 -------------
    const int nB    = tid & 255;
    const int halfB = tid >> 8;           // 0 or 1
    // LDS aperture: flat addr[31:0] == LDS byte offset
    const unsigned ldsB =
        (unsigned)(uintptr_t)&Bsh[nB * LDB + halfB * 16];

    v8f acc[2][4] = {};

    for (int kc0 = 0; kc0 < KC; kc0 += 32) {
        __syncthreads();  // previous slice's WMMA reads of LDS are done

        // ---------------- stage B: async filter slice -> LDS (col-major) ---
        {
            const unsigned short* src =
                filtT + (size_t)nB * KC + kc0 + halfB * 16;
            // two 16B async copies: global -> LDS, tracked by ASYNCcnt
            asm volatile(
                "global_load_async_to_lds_b128 %0, %2, off\n\t"
                "global_load_async_to_lds_b128 %1, %3, off"
                :
                : "v"(ldsB), "v"(ldsB + 16),
                  "v"((unsigned long long)(uintptr_t)src),
                  "v"((unsigned long long)(uintptr_t)(src + 8))
                : "memory");
            if (kc0 + 32 < KC)                          // hint next slice
                __builtin_prefetch(src + 32, 0, 1);
        }

        // ---------------- stage A: modulated bilinear sample -> LDS --------
        {
            const int k     = kc0 >> 7;       // tap index 0..8
            const int cBase = kc0 & 127;      // channel base within slice row
            const float o1 = orow[k];
            const float o2 = orow[9 + k];
            const float mv = orow[18 + k];
            const float msk = 1.0f / (1.0f + __expf(-mv));
            const int ky = k / 3, kx = k - ky * 3;
            const float py = (float)(h - 1 + ky) + o1;
            const float px = (float)(w - 1 + kx) + o2;
            const float y0 = floorf(py), x0 = floorf(px);
            const float wy1 = py - y0, wy0 = 1.0f - wy1;
            const float wx1 = px - x0, wx0 = 1.0f - wx1;
            const int iy0 = (int)y0, ix0 = (int)x0;

            const int c0 = cBase + cg * 8;
            float v[8];
            #pragma unroll
            for (int i = 0; i < 8; ++i) v[i] = 0.0f;

            const int   dys[4] = {0, 0, 1, 1};
            const int   dxs[4] = {0, 1, 0, 1};
            const float wts[4] = {wy0 * wx0, wy0 * wx1, wy1 * wx0, wy1 * wx1};
            #pragma unroll
            for (int cn = 0; cn < 4; ++cn) {
                const int yy = iy0 + dys[cn];
                const int xx = ix0 + dxs[cn];
                if (yy >= 0 && yy < HH && xx >= 0 && xx < WW) {
                    const float wt = wts[cn];
                    const float* p =
                        in + ((size_t)((b * HH + yy) * WW + xx)) * CC + c0;
                    const float4 q0 = *(const float4*)(p);
                    const float4 q1 = *(const float4*)(p + 4);
                    v[0] = fmaf(wt, q0.x, v[0]); v[1] = fmaf(wt, q0.y, v[1]);
                    v[2] = fmaf(wt, q0.z, v[2]); v[3] = fmaf(wt, q0.w, v[3]);
                    v[4] = fmaf(wt, q1.x, v[4]); v[5] = fmaf(wt, q1.y, v[5]);
                    v[6] = fmaf(wt, q1.z, v[6]); v[7] = fmaf(wt, q1.w, v[7]);
                }
            }
            unsigned short* dst = &Ash[rowL * LDA + cg * 8];
            #pragma unroll
            for (int i = 0; i < 8; ++i) dst[i] = f2bf(v[i] * msk);
        }

        // wait for async B copy, then make both tiles visible to all waves
        asm volatile("s_wait_asynccnt 0x0" ::: "memory");
        __syncthreads();

        // ---------------- WMMA over the 32-deep slice ----------------------
        // A frag (ISA 7.12.2, 16-bit A 16x32): lanes<16 hold K 0..7 & 16..23,
        // lanes>=16 hold K 8..15 & 24..31, M = lane&15.
        FragBF fa[2];
        const int koff = (lane < 16) ? 0 : 8;
        #pragma unroll
        for (int i = 0; i < 2; ++i) {
            const unsigned short* ap =
                &Ash[(wm * 32 + i * 16 + (lane & 15)) * LDA + koff];
            fa[i].q[0] = *(const uint4*)(ap);
            fa[i].q[1] = *(const uint4*)(ap + 16);
        }
        // B frag: lanes<16 hold K 0..7 & 16..23, lanes>=16 hold K 8..15 & 24..31,
        // N = lane&15; from column-major LDS.
        FragBF fb[4];
        const int kbase = (lane < 16) ? 0 : 8;
        #pragma unroll
        for (int j = 0; j < 4; ++j) {
            const unsigned short* bp =
                &Bsh[(wn * 64 + j * 16 + (lane & 15)) * LDB + kbase];
            fb[j].q[0] = *(const uint4*)(bp);
            fb[j].q[1] = *(const uint4*)(bp + 16);
        }
        #pragma unroll
        for (int i = 0; i < 2; ++i)
            #pragma unroll
            for (int j = 0; j < 4; ++j)
                acc[i][j] = __builtin_amdgcn_wmma_f32_16x16x32_bf16(
                    false, fa[i].v, false, fb[j].v,
                    (short)0, acc[i][j], false, false);
    }

    // ---------------- store D: VGPR r -> M = r (+8 for hi lanes) ------------
    const int mBase = blockM + wm * 32;
    const int mHi   = (lane < 16) ? 0 : 8;
    const int nLane = lane & 15;
    #pragma unroll
    for (int i = 0; i < 2; ++i) {
        #pragma unroll
        for (int j = 0; j < 4; ++j) {
            const int n = wn * 64 + j * 16 + nLane;
            #pragma unroll
            for (int r = 0; r < 8; ++r) {
                const int m = mBase + i * 16 + r + mHi;
                out[(size_t)m * FF + n] = acc[i][j][r];
            }
        }
    }
}

// ---------------------------------------------------------------------------
// Host launcher
// ---------------------------------------------------------------------------
extern "C" void kernel_launch(void* const* d_in, const int* in_sizes, int n_in,
                              void* d_out, int out_size, void* d_ws, size_t ws_size,
                              hipStream_t stream) {
    const float* in    = (const float*)d_in[0];   // [8,64,64,128]
    const float* okern = (const float*)d_in[1];   // [3,3,128,27]
    const float* obias = (const float*)d_in[2];   // [27]
    const float* filt  = (const float*)d_in[3];   // [3,3,128,256]
    float* out = (float*)d_out;                   // [8,64,64,256]

    char* wsb = (char*)d_ws;
    float* ofs = (float*)wsb;                                   // 32768*27 f32
    unsigned short* filtT =
        (unsigned short*)(wsb + (size_t)NPIX * 27 * sizeof(float)); // 256*1152 bf16

    dcn_offset_conv<<<(NPIX * 27 + 255) / 256, 256, 0, stream>>>(
        in, okern, obias, ofs);
    dcn_filt_cvt<<<(FF * KC + 255) / 256, 256, 0, stream>>>(filt, filtT);
    dcn_gemm<<<NPIX / 128, 512, 0, stream>>>(in, ofs, filtT, out);
}
